// SparseMoe_22153441313344
// MI455X (gfx1250) — compile-verified, matrix operand order checked by reading, not run on hardware
//
#include <hip/hip_runtime.h>
#include <hip/hip_bf16.h>
#include <stdint.h>

// ---------------------------------------------------------------------------
// Sparse MoE (top-2 of 8 experts), B=4 S=2048 D=1024 E=8, fp32 in/out.
//   K0 : zero out + bucket counters
//   K0b: convert x and expert_w to bf16 once (workspace)
//   K1 : router -> logits out, softmax, top-2, renorm, bucket append
//   K2 : gathered 128x64 bf16 WMMA GEMM, async-to-LDS double buffered,
//        weighted atomic scatter (exactly 2 commutative adds/elem).
// ---------------------------------------------------------------------------

#define TT 8192      // tokens = B*S
#define DD 1024      // model dim
#define EE 8         // experts

typedef __attribute__((ext_vector_type(16))) __bf16 v16bf;
typedef __attribute__((ext_vector_type(8)))  float  v8f;

union Frag {
    unsigned int u[8];
    v16bf v;
};

// round-to-nearest-even fp32 -> bf16, packed pair into one dword
__device__ __forceinline__ unsigned int pack_bf16(float lo, float hi) {
    unsigned int a = __float_as_uint(lo);
    unsigned int b = __float_as_uint(hi);
    a = (a + 0x7FFFu + ((a >> 16) & 1u)) >> 16;
    b = (b + 0x7FFFu + ((b >> 16) & 1u)) >> 16;
    return (a & 0xFFFFu) | (b << 16);
}

// low 32 bits of a generic pointer to __shared__ == LDS byte offset
__device__ __forceinline__ unsigned lds_off(const void* p) {
    return (unsigned)(unsigned long long)p;
}

// CDNA5 async copy: 16B global -> LDS per lane, tracked by ASYNCcnt
__device__ __forceinline__ void async_cp16(unsigned lds, unsigned long long gaddr) {
    asm volatile("global_load_async_to_lds_b128 %0, %1, off"
                 :: "v"(lds), "v"(gaddr) : "memory");
}

// ---------------------------------------------------------------------------
// Kernel 0: zero the dense output region and the expert bucket counters.
// ---------------------------------------------------------------------------
__global__ __launch_bounds__(256) void moe_zero(float* __restrict__ out,
                                                int* __restrict__ counts) {
    size_t i = ((size_t)blockIdx.x * 256 + threadIdx.x) * 4;
    if (i < (size_t)TT * DD) {
        float4 z = make_float4(0.f, 0.f, 0.f, 0.f);
        *(float4*)(out + i) = z;
    }
    if (blockIdx.x == 0 && threadIdx.x < EE) counts[threadIdx.x] = 0;
}

// ---------------------------------------------------------------------------
// Kernel 0b: one-shot fp32 -> packed bf16 conversion (8 floats / thread).
// ---------------------------------------------------------------------------
__global__ __launch_bounds__(256) void moe_cvt_bf16(const float* __restrict__ src,
                                                    unsigned* __restrict__ dst,
                                                    size_t n8) {
    size_t i = (size_t)blockIdx.x * 256 + threadIdx.x;
    if (i >= n8) return;
    const float4* s = (const float4*)(src + i * 8);
    float4 a = s[0], b = s[1];
    *(uint4*)(dst + i * 4) = make_uint4(pack_bf16(a.x, a.y), pack_bf16(a.z, a.w),
                                        pack_bf16(b.x, b.y), pack_bf16(b.z, b.w));
}

// ---------------------------------------------------------------------------
// Kernel 1: router. One wave per token. fp32 logits/softmax/top-2/renorm.
// ---------------------------------------------------------------------------
__global__ __launch_bounds__(256) void moe_router(
        const float* __restrict__ x,
        const float* __restrict__ gate_w,
        const float* __restrict__ gate_b,
        float* __restrict__ logits_out,   // [TT, EE]
        int*   __restrict__ counts,       // [EE]
        int*   __restrict__ bidx,         // [EE, TT]
        float* __restrict__ bwt)          // [EE, TT]
{
    const int wave = threadIdx.x >> 5;
    const int lane = threadIdx.x & 31;
    const int t = blockIdx.x * 8 + wave;
    if (t >= TT) return;

    const float* xr = x + (size_t)t * DD;
    float xv[32];
#pragma unroll
    for (int i = 0; i < 32; ++i) xv[i] = xr[lane + (i << 5)];

    float logit[EE];
#pragma unroll
    for (int e = 0; e < EE; ++e) {
        const float* gw = gate_w + e * DD;
        float s = 0.f;
#pragma unroll
        for (int i = 0; i < 32; ++i) s = fmaf(xv[i], gw[lane + (i << 5)], s);
#pragma unroll
        for (int m = 16; m; m >>= 1) s += __shfl_xor(s, m, 32);
        logit[e] = s + gate_b[e];
    }

    if (lane == 0) {
#pragma unroll
        for (int e = 0; e < EE; ++e) logits_out[(size_t)t * EE + e] = logit[e];
        int i0 = 0; float l0 = logit[0];
#pragma unroll
        for (int e = 1; e < EE; ++e)
            if (logit[e] > l0) { l0 = logit[e]; i0 = e; }
        int i1 = -1; float l1 = -3.4e38f;
#pragma unroll
        for (int e = 0; e < EE; ++e)
            if (e != i0 && logit[e] > l1) { l1 = logit[e]; i1 = e; }
        float r  = __expf(l1 - l0);
        float w0 = 1.0f / (1.0f + r);
        float w1 = r / (1.0f + r);
        int s0 = atomicAdd(&counts[i0], 1);
        bidx[i0 * TT + s0] = t; bwt[i0 * TT + s0] = w0;
        int s1 = atomicAdd(&counts[i1], 1);
        bidx[i1 * TT + s1] = t; bwt[i1 * TT + s1] = w1;
    }
}

// ---------------------------------------------------------------------------
// Kernel 2: gathered expert GEMM, bf16 WMMA, fp32 accumulate.
// Block = 256 threads = 8 waves. Tile: 128 tokens x 64 outputs, K-chunk 64.
// Wave w owns m-subtile w (16 rows) and all four 16-col n-subtiles
// (A fragment reused 4x). Double-buffered LDS filled by async-to-LDS.
// grid = (DD/64, TT/128 worst case, EE); early-exit on bucket count.
// ---------------------------------------------------------------------------
__global__ __launch_bounds__(256) void moe_gemm(
        const unsigned short* __restrict__ xb,   // [TT, DD]  bf16
        const unsigned short* __restrict__ wb,   // [EE, DD(out), DD(in)] bf16
        const float* __restrict__ expert_b,      // [EE, DD]
        const int*   __restrict__ counts,
        const int*   __restrict__ bidx,
        const float* __restrict__ bwt,
        float* __restrict__ out)                 // [TT, DD]
{
    const int e   = blockIdx.z;
    const int cnt = counts[e];
    const int m0  = blockIdx.y * 128;
    if (m0 >= cnt) return;
    const int n0  = blockIdx.x * 64;

    // 36-dword row stride: (r*36 + c) mod 64 distinct for r=0..15 -> no bank
    // conflicts on ds_load_b128 fragment reads.
    __shared__ unsigned int As[2][128][36];
    __shared__ unsigned int Bs[2][64][36];
    __shared__ int   idx_s[128];
    __shared__ float wt_s[128];

    const int tid = threadIdx.x;
    if (tid < 128) {
        int g = m0 + tid;
        bool valid = g < cnt;
        idx_s[tid] = valid ? bidx[e * TT + g] : -1;
        wt_s[tid]  = valid ? bwt[e * TT + g] : 0.f;
    }
    __syncthreads();

    // ---- staging geometry (async 16B copies) ----
    // A: 128 rows x 64 bf16 = 16KB; 2 threads/row, 4 x 16B each
    const int rA = tid >> 1, qA = tid & 1;
    const int tokA = max(idx_s[rA], 0);            // clamp; masked in epilogue
    const unsigned long long gA =
        (unsigned long long)(const void*)(xb + (size_t)tokA * DD + qA * 32);
    const unsigned ldsA0 = lds_off(&As[0][rA][qA * 16]);
    const unsigned ldsA1 = lds_off(&As[1][rA][qA * 16]);
    // B: 64 rows x 64 bf16 = 8KB; 4 threads/row, 2 x 16B each
    const int rB = tid >> 2, qB = tid & 3;
    const unsigned long long gB =
        (unsigned long long)(const void*)(wb + ((size_t)e * DD + n0 + rB) * DD + qB * 16);
    const unsigned ldsB0 = lds_off(&Bs[0][rB][qB * 8]);
    const unsigned ldsB1 = lds_off(&Bs[1][rB][qB * 8]);

    // ---- fragment geometry ----
    const int wid  = tid >> 5;            // m-subtile 0..7
    const int lane = tid & 31;
    const int rowA   = (lane & 15) + wid * 16;
    const int rowB   = lane & 15;
    const int khalf4 = (lane >> 4) * 4;   // bf16-pair offset for lane half

    v8f acc[4] = {{}, {}, {}, {}};

    // issue one 64-wide K chunk into buffer `buf` (6 async ops per wave)
    auto issue = [&](int k0, int buf) {
        unsigned long long goA = gA + (unsigned long long)k0 * 2;
        unsigned long long goB = gB + (unsigned long long)k0 * 2;
        unsigned la = buf ? ldsA1 : ldsA0;
        unsigned lb = buf ? ldsB1 : ldsB0;
#pragma unroll
        for (int j = 0; j < 4; ++j) async_cp16(la + j * 16, goA + j * 16);
#pragma unroll
        for (int j = 0; j < 2; ++j) async_cp16(lb + j * 16, goB + j * 16);
    };

    issue(0, 0);

    for (int c = 0; c < DD / 64; ++c) {
        const int buf = c & 1;
        const bool has_next = (c + 1) < (DD / 64);
        if (has_next) {
            issue((c + 1) * 64, buf ^ 1);
            asm volatile("s_wait_asynccnt 0x6" ::: "memory");  // chunk c done
        } else {
            asm volatile("s_wait_asynccnt 0x0" ::: "memory");
        }
        __syncthreads();   // all waves' chunk-c data visible

#pragma unroll
        for (int kk = 0; kk < 64; kk += 32) {
            const int base = (kk >> 1) + khalf4;   // bf16-pair index
            Frag a;
#pragma unroll
            for (int j = 0; j < 4; ++j) {
                a.u[j]     = As[buf][rowA][base + j];
                a.u[4 + j] = As[buf][rowA][base + 8 + j];
            }
#pragma unroll
            for (int nt = 0; nt < 4; ++nt) {
                Frag b;
#pragma unroll
                for (int j = 0; j < 4; ++j) {
                    b.u[j]     = Bs[buf][nt * 16 + rowB][base + j];
                    b.u[4 + j] = Bs[buf][nt * 16 + rowB][base + 8 + j];
                }
                acc[nt] = __builtin_amdgcn_wmma_f32_16x16x32_bf16(
                              false, a.v, false, b.v, (short)0, acc[nt],
                              false, false);
            }
        }
        __syncthreads();   // write-after-read fence before buf is refilled
    }

    // ---- epilogue: weighted scatter-add (2 commutative adds per element) ----
    const int colL = lane & 15;
    const int half = lane >> 4;
#pragma unroll
    for (int nt = 0; nt < 4; ++nt) {
        const int cg = n0 + nt * 16 + colL;
        const float bias = expert_b[e * DD + cg];
#pragma unroll
        for (int i = 0; i < 8; ++i) {
            const int rloc = wid * 16 + half * 8 + i;
            if (m0 + rloc < cnt) {
                const int   t = idx_s[rloc];
                const float w = wt_s[rloc];
                atomicAdd(&out[(size_t)t * DD + cg], w * (acc[nt][i] + bias));
            }
        }
    }
}

// ---------------------------------------------------------------------------
extern "C" void kernel_launch(void* const* d_in, const int* in_sizes, int n_in,
                              void* d_out, int out_size, void* d_ws, size_t ws_size,
                              hipStream_t stream) {
    const float* x        = (const float*)d_in[0];  // [B,S,D] fp32
    const float* gate_w   = (const float*)d_in[1];  // [E,D]
    const float* gate_b   = (const float*)d_in[2];  // [E]
    const float* expert_w = (const float*)d_in[3];  // [E,D,D]
    const float* expert_b = (const float*)d_in[4];  // [E,D]

    float* out    = (float*)d_out;                  // [T,D]
    float* logits = out + (size_t)TT * DD;          // [T,E] (tuple output #2)

    // workspace layout
    char* ws = (char*)d_ws;
    int*   counts = (int*)ws;                                  // 256 B
    int*   bidx   = (int*)(ws + 256);                          // E*T*4 = 256 KB
    float* bwt    = (float*)(ws + 256 + (size_t)EE * TT * 4);  // 256 KB
    unsigned short* xbuf = (unsigned short*)(ws + (1u << 20));            // 16 MB
    unsigned short* wbuf = (unsigned short*)(ws + (1u << 20) + (16u << 20)); // 16 MB

    // K0: zero dense output + bucket counters
    moe_zero<<<(TT * (size_t)DD) / (256 * 4), 256, 0, stream>>>(out, counts);

    // K0b: one-shot bf16 conversion of activations and expert weights
    const size_t n8 = (size_t)TT * DD / 8;     // == E*D*D/8 as well
    moe_cvt_bf16<<<(unsigned)(n8 / 256), 256, 0, stream>>>(x, (unsigned*)xbuf, n8);
    moe_cvt_bf16<<<(unsigned)(n8 / 256), 256, 0, stream>>>(expert_w, (unsigned*)wbuf, n8);

    // K1: router
    moe_router<<<TT / 8, 256, 0, stream>>>(x, gate_w, gate_b,
                                           logits, counts, bidx, bwt);

    // K2: gathered bf16 WMMA GEMM + weighted combine
    dim3 grid(DD / 64, TT / 128, EE);
    moe_gemm<<<grid, 256, 0, stream>>>(xbuf, wbuf, expert_b,
                                       counts, bidx, bwt, out);
}
